// mySelfAttention_23484881175168
// MI455X (gfx1250) — matrix operand order repaired and checked by reading, not compiled, not run on hardware
//
#include <hip/hip_runtime.h>
#include <hip/hip_bf16.h>

#define H_DIM 1024
#define NH    16
#define HD    64
#define BATCH 4
#define SEQ   2048
#define M_TOT (BATCH * SEQ)   // 8192
#define LN_EPS 1e-5f

typedef __attribute__((ext_vector_type(16))) __bf16 v16bf;
typedef __attribute__((ext_vector_type(8)))  float  v8f;

union Frag16 { v16bf v; uint4 q[2]; };

__device__ __forceinline__ v8f v8f_zero() {
  v8f z = {0.f, 0.f, 0.f, 0.f, 0.f, 0.f, 0.f, 0.f};
  return z;
}

__device__ __forceinline__ v8f wmma_bf16(const Frag16& a, const Frag16& b, v8f c) {
  // D = A(16x32 bf16) * B(32x16 bf16) + C(16x16 f32)
  return __builtin_amdgcn_wmma_f32_16x16x32_bf16(
      /*neg_a=*/false, a.v, /*neg_b=*/false, b.v,
      /*c_mod=*/(short)0, c, /*reuse_a=*/false, /*reuse_b=*/false);
}

// fp32 -> bf16 (round to nearest even), raw ushort storage.
__device__ __forceinline__ unsigned short to_bf16(float f) {
  union { float f; unsigned u; } a; a.f = f;
  unsigned r = a.u + 0x7FFFu + ((a.u >> 16) & 1u);
  return (unsigned short)(r >> 16);
}

__device__ __forceinline__ void lds_fence_wave() {
  // order our own DS stores before subsequent DS loads (cross-lane exchange
  // within one wave, lanes in lockstep); CDNA5 split counter.
  asm volatile("s_wait_dscnt 0" ::: "memory");
}

// ---------------------------------------------------------------------------
// Kernel 1: LayerNorm(x) -> bf16 h, one 256-thread block per row of 1024
// ---------------------------------------------------------------------------
__global__ __launch_bounds__(256) void ln_bf16_kernel(
    const float* __restrict__ x, const float* __restrict__ g,
    const float* __restrict__ beta, unsigned short* __restrict__ h) {
  const int row = blockIdx.x;
  const int tid = threadIdx.x;
  const float4 v = ((const float4*)(x + (size_t)row * H_DIM))[tid];

  float s = v.x + v.y + v.z + v.w;
  float q = v.x * v.x + v.y * v.y + v.z * v.z + v.w * v.w;
#pragma unroll
  for (int o = 16; o >= 1; o >>= 1) {
    s += __shfl_xor(s, o, 32);
    q += __shfl_xor(q, o, 32);
  }
  __shared__ float reds[8], redq[8];
  const int wid = tid >> 5, lane = tid & 31;
  if (lane == 0) { reds[wid] = s; redq[wid] = q; }
  __syncthreads();
  s = 0.f; q = 0.f;
#pragma unroll
  for (int i = 0; i < 8; ++i) { s += reds[i]; q += redq[i]; }

  const float mu   = s * (1.f / H_DIM);
  const float var  = q * (1.f / H_DIM) - mu * mu;
  const float rstd = rsqrtf(var + LN_EPS);

  const float4 gg = ((const float4*)g)[tid];
  const float4 bb = ((const float4*)beta)[tid];
  ushort4 o = make_ushort4(to_bf16((v.x - mu) * rstd * gg.x + bb.x),
                           to_bf16((v.y - mu) * rstd * gg.y + bb.y),
                           to_bf16((v.z - mu) * rstd * gg.z + bb.z),
                           to_bf16((v.w - mu) * rstd * gg.w + bb.w));
  ((ushort4*)(h + (size_t)row * H_DIM))[tid] = o;
}

// ---------------------------------------------------------------------------
// Kernel 2: fp32 -> bf16 bulk convert (weights), 4 elements / thread
// ---------------------------------------------------------------------------
__global__ __launch_bounds__(256) void cvt_bf16_kernel(
    const float* __restrict__ in, unsigned short* __restrict__ out) {
  const int i = blockIdx.x * 256 + threadIdx.x;
  const float4 v = ((const float4*)in)[i];
  ((ushort4*)out)[i] = make_ushort4(to_bf16(v.x), to_bf16(v.y),
                                    to_bf16(v.z), to_bf16(v.w));
}

// ---------------------------------------------------------------------------
// Kernel 3: QKV GEMM.  C[8192 x 3072] = h @ W3^T  (W3 = [Wq;Wk;Wv]).
// Block tile 128x128, K-step 32, LDS double-buffered (1 barrier / step),
// 8 waves x (32x64) wave tiles (2x4 of 16x16x32 WMMAs).
// Epilogue scatters into head-major q/k and d-major (transposed) v.
// ---------------------------------------------------------------------------
__global__ __launch_bounds__(256) void gemm_qkv_kernel(
    const unsigned short* __restrict__ A,    // h_bf16 [8192][1024]
    const unsigned short* __restrict__ W3,   // [3072][1024] bf16
    const float* __restrict__ bq, const float* __restrict__ bk,
    const float* __restrict__ bv,
    unsigned short* __restrict__ qo,         // [B][NH][SEQ][HD]
    unsigned short* __restrict__ ko,         // [B][NH][SEQ][HD]
    unsigned short* __restrict__ vo) {       // [B][NH][HD][SEQ] (transposed)
  constexpr int LDL = 40;                    // 80B pitch: conflict-light
  __shared__ __align__(16) unsigned short As[2][128 * LDL];
  __shared__ __align__(16) unsigned short Bs[2][128 * LDL];

  const int tid = threadIdx.x;
  const int lane = tid & 31, wid = tid >> 5;
  const int half = lane >> 4, l = lane & 15;
  const int wm = wid >> 1, wn = wid & 1;
  const int mBase = blockIdx.y * 128;
  const int nBase = blockIdx.x * 128;

  // this thread's two 16B staging slots (512 chunks over 256 threads)
  const int row0 = (tid + 0)   >> 2, kc0 = ((tid + 0)   & 3) * 8;
  const int row1 = (tid + 256) >> 2, kc1 = ((tid + 256) & 3) * 8;
  const unsigned short* gA0 = A  + (size_t)(mBase + row0) * H_DIM + kc0;
  const unsigned short* gA1 = A  + (size_t)(mBase + row1) * H_DIM + kc1;
  const unsigned short* gB0 = W3 + (size_t)(nBase + row0) * H_DIM + kc0;
  const unsigned short* gB1 = W3 + (size_t)(nBase + row1) * H_DIM + kc1;

  v8f acc[2][4];
#pragma unroll
  for (int s = 0; s < 2; ++s)
#pragma unroll
    for (int t = 0; t < 4; ++t) acc[s][t] = v8f_zero();

  // prologue: stage K-slab 0 into buffer 0
  *(uint4*)&As[0][row0 * LDL + kc0] = *(const uint4*)gA0;
  *(uint4*)&As[0][row1 * LDL + kc1] = *(const uint4*)gA1;
  *(uint4*)&Bs[0][row0 * LDL + kc0] = *(const uint4*)gB0;
  *(uint4*)&Bs[0][row1 * LDL + kc1] = *(const uint4*)gB1;
  __syncthreads();

  int buf = 0;
  for (int k0 = 0; k0 < H_DIM; k0 += 32, buf ^= 1) {
    const bool more = (k0 + 32 < H_DIM);
    uint4 na0, na1, nb0, nb1;
    if (more) {                              // next slab -> registers
      na0 = *(const uint4*)(gA0 + k0 + 32);
      na1 = *(const uint4*)(gA1 + k0 + 32);
      nb0 = *(const uint4*)(gB0 + k0 + 32);
      nb1 = *(const uint4*)(gB1 + k0 + 32);
      if (k0 + 64 < H_DIM) {                 // prefetch slab after that
        __builtin_prefetch(gA0 + k0 + 64, 0, 3);
        __builtin_prefetch(gB0 + k0 + 64, 0, 3);
      }
    }

    Frag16 a[2], b[4];
#pragma unroll
    for (int s = 0; s < 2; ++s) {            // A 16x32: lane<16 K0..7|16..23
      const int r = wm * 32 + s * 16 + l;
      a[s].q[0] = *(const uint4*)&As[buf][r * LDL + half * 8];
      a[s].q[1] = *(const uint4*)&As[buf][r * LDL + 16 + half * 8];
    }
#pragma unroll
    for (int t = 0; t < 4; ++t) {            // B 32x16: lane<16 K0..7|16..23
      const int n = wn * 64 + t * 16 + l;
      b[t].q[0] = *(const uint4*)&Bs[buf][n * LDL + half * 8];
      b[t].q[1] = *(const uint4*)&Bs[buf][n * LDL + 16 + half * 8];
    }
#pragma unroll
    for (int s = 0; s < 2; ++s)
#pragma unroll
      for (int t = 0; t < 4; ++t) acc[s][t] = wmma_bf16(a[s], b[t], acc[s][t]);

    if (more) {                              // commit next slab to other buf
      *(uint4*)&As[buf ^ 1][row0 * LDL + kc0] = na0;
      *(uint4*)&As[buf ^ 1][row1 * LDL + kc1] = na1;
      *(uint4*)&Bs[buf ^ 1][row0 * LDL + kc0] = nb0;
      *(uint4*)&Bs[buf ^ 1][row1 * LDL + kc1] = nb1;
    }
    __syncthreads();
  }

  const int proj = nBase >> 10;              // 0=q 1=k 2=v
  const float* bias = (proj == 0) ? bq : ((proj == 1) ? bk : bv);
  unsigned short* outp = (proj == 0) ? qo : ((proj == 1) ? ko : vo);
#pragma unroll
  for (int s = 0; s < 2; ++s) {
#pragma unroll
    for (int t = 0; t < 4; ++t) {
      const int col = nBase + wn * 64 + t * 16 + l;
      const int nn = col & 1023;
      const float bb = bias[nn];
      const int head = nn >> 6, d = nn & 63;
#pragma unroll
      for (int r = 0; r < 8; ++r) {          // C layout: lane row = 8*half+r
        const int m = mBase + wm * 32 + s * 16 + half * 8 + r;
        const int bidx = m >> 11, tt = m & (SEQ - 1);
        const float val = acc[s][t][r] + bb;
        const size_t off = (proj < 2)
            ? ((size_t)(bidx * NH + head) * SEQ + tt) * HD + d
            : ((size_t)(bidx * NH + head) * HD + d) * SEQ + tt;
        outp[off] = to_bf16(val);
      }
    }
  }
}

// ---------------------------------------------------------------------------
// Kernel 4: flash attention.  grid = (SEQ/128, B*NH); 8 waves x 16 q-rows.
// 64-key tiles: S = Q K^T via 8 WMMAs, online softmax with max-only in-loop
// reduction (row-sum kept as per-lane partial, reduced once at the end),
// P transposed through per-wave LDS, O += P V via 8 WMMAs.
// ---------------------------------------------------------------------------
__global__ __launch_bounds__(256) void attn_kernel(
    const unsigned short* __restrict__ q,    // [B][NH][SEQ][HD]
    const unsigned short* __restrict__ k,    // [B][NH][SEQ][HD]
    const unsigned short* __restrict__ vT,   // [B][NH][HD][SEQ]
    const int* __restrict__ amask,           // [B][SEQ]
    unsigned short* __restrict__ c) {        // [8192][1024] bf16
  constexpr int PLD = 72;                    // 64 keys + 8 pad (144B pitch)
  __shared__ __align__(16) unsigned short Pst[8][16 * PLD];

  const int tid = threadIdx.x;
  const int lane = tid & 31, wid = tid >> 5;
  const int half = lane >> 4, l = lane & 15;
  const int bh = blockIdx.y;
  const int b = bh >> 4, head = bh & 15;
  const int qt0 = blockIdx.x * 128 + wid * 16;

  const unsigned short* qp = q + ((size_t)bh * SEQ + qt0) * HD;
  const unsigned short* kp = k + (size_t)bh * SEQ * HD;
  const unsigned short* vp = vT + (size_t)bh * HD * SEQ;
  const int* mrow = amask + b * SEQ;
  unsigned short* pw = &Pst[wid][0];

  Frag16 Qa[2];                              // Q 16x64 as two 16x32 A-frags
#pragma unroll
  for (int s = 0; s < 2; ++s) {
    const unsigned short* qr = qp + (size_t)l * HD + s * 32 + half * 8;
    Qa[s].q[0] = *(const uint4*)qr;
    Qa[s].q[1] = *(const uint4*)(qr + 16);
  }

  v8f O[4];
#pragma unroll
  for (int t = 0; t < 4; ++t) O[t] = v8f_zero();
  float mrun[8], lrun[8];
#pragma unroll
  for (int r = 0; r < 8; ++r) { mrun[r] = -1e30f; lrun[r] = 0.f; }

  const float scale = 0.125f;                // 1/sqrt(64)

  for (int kt = 0; kt < SEQ; kt += 64) {
    v8f S[4];
#pragma unroll
    for (int g = 0; g < 4; ++g) S[g] = v8f_zero();
#pragma unroll
    for (int g = 0; g < 4; ++g) {            // keys [kt+16g, kt+16g+16)
      Frag16 Kb;
      const unsigned short* kr = kp + (size_t)(kt + g * 16 + l) * HD + half * 8;
      Kb.q[0] = *(const uint4*)kr;  Kb.q[1] = *(const uint4*)(kr + 16);
      S[g] = wmma_bf16(Qa[0], Kb, S[g]);
      Kb.q[0] = *(const uint4*)(kr + 32); Kb.q[1] = *(const uint4*)(kr + 48);
      S[g] = wmma_bf16(Qa[1], Kb, S[g]);
    }

    int mk[4];
#pragma unroll
    for (int g = 0; g < 4; ++g) mk[g] = mrow[kt + g * 16 + l];

#pragma unroll
    for (int r = 0; r < 8; ++r) {
      float sv[4];
#pragma unroll
      for (int g = 0; g < 4; ++g) sv[g] = mk[g] ? S[g][r] * scale : -1e30f;
      float mx = fmaxf(fmaxf(sv[0], sv[1]), fmaxf(sv[2], sv[3]));
#pragma unroll
      for (int o = 1; o < 16; o <<= 1) mx = fmaxf(mx, __shfl_xor(mx, o, 32));
      const float mnew = fmaxf(mrun[r], mx);
      const float alpha = __expf(mrun[r] - mnew);
      mrun[r] = mnew;
      float ps = 0.f;
#pragma unroll
      for (int g = 0; g < 4; ++g) {
        const float p = __expf(sv[g] - mnew);
        ps += p;
        pw[(half * 8 + r) * PLD + g * 16 + l] = to_bf16(p);
      }
      lrun[r] = lrun[r] * alpha + ps;        // per-lane partial row sum
#pragma unroll
      for (int t = 0; t < 4; ++t) O[t][r] *= alpha;
    }
    lds_fence_wave();

    Frag16 Pa[2];                            // P 16x64 as two A-fragments
    Pa[0].q[0] = *(const uint4*)&pw[l * PLD + half * 8];
    Pa[0].q[1] = *(const uint4*)&pw[l * PLD + 16 + half * 8];
    Pa[1].q[0] = *(const uint4*)&pw[l * PLD + 32 + half * 8];
    Pa[1].q[1] = *(const uint4*)&pw[l * PLD + 48 + half * 8];
#pragma unroll
    for (int t = 0; t < 4; ++t) {            // V^T rows contiguous in keys
      Frag16 Vb;
      const unsigned short* vr = vp + (size_t)(t * 16 + l) * SEQ + kt + half * 8;
      Vb.q[0] = *(const uint4*)vr;
      Vb.q[1] = *(const uint4*)(vr + 16);
      O[t] = wmma_bf16(Pa[0], Vb, O[t]);
      Vb.q[0] = *(const uint4*)(vr + 32);
      Vb.q[1] = *(const uint4*)(vr + 48);
      O[t] = wmma_bf16(Pa[1], Vb, O[t]);
    }
  }

#pragma unroll
  for (int r = 0; r < 8; ++r) {
    float rs = lrun[r];                      // reduce partial sums once
#pragma unroll
    for (int o = 1; o < 16; o <<= 1) rs += __shfl_xor(rs, o, 32);
    const float inv = (rs > 0.f) ? (1.f / rs) : 0.f;
    const size_t row =
        (size_t)(b * SEQ + qt0 + half * 8 + r) * H_DIM + head * HD;
#pragma unroll
    for (int t = 0; t < 4; ++t)
      c[row + t * 16 + l] = to_bf16(O[t][r] * inv);
  }
}

// ---------------------------------------------------------------------------
// Kernel 5: output projection + bias + residual, fp32 out.
// out[8192 x 1024] = c_bf16 @ Wo^T + bo + x   (same double-buffered loop)
// ---------------------------------------------------------------------------
__global__ __launch_bounds__(256) void gemm_out_kernel(
    const unsigned short* __restrict__ A,    // c_bf16 [8192][1024]
    const unsigned short* __restrict__ W,    // Wo bf16 [1024][1024]
    const float* __restrict__ bo, const float* __restrict__ x,
    float* __restrict__ out) {
  constexpr int LDL = 40;
  __shared__ __align__(16) unsigned short As[2][128 * LDL];
  __shared__ __align__(16) unsigned short Bs[2][128 * LDL];

  const int tid = threadIdx.x;
  const int lane = tid & 31, wid = tid >> 5;
  const int half = lane >> 4, l = lane & 15;
  const int wm = wid >> 1, wn = wid & 1;
  const int mBase = blockIdx.y * 128;
  const int nBase = blockIdx.x * 128;

  const int row0 = (tid + 0)   >> 2, kc0 = ((tid + 0)   & 3) * 8;
  const int row1 = (tid + 256) >> 2, kc1 = ((tid + 256) & 3) * 8;
  const unsigned short* gA0 = A + (size_t)(mBase + row0) * H_DIM + kc0;
  const unsigned short* gA1 = A + (size_t)(mBase + row1) * H_DIM + kc1;
  const unsigned short* gB0 = W + (size_t)(nBase + row0) * H_DIM + kc0;
  const unsigned short* gB1 = W + (size_t)(nBase + row1) * H_DIM + kc1;

  v8f acc[2][4];
#pragma unroll
  for (int s = 0; s < 2; ++s)
#pragma unroll
    for (int t = 0; t < 4; ++t) acc[s][t] = v8f_zero();

  *(uint4*)&As[0][row0 * LDL + kc0] = *(const uint4*)gA0;
  *(uint4*)&As[0][row1 * LDL + kc1] = *(const uint4*)gA1;
  *(uint4*)&Bs[0][row0 * LDL + kc0] = *(const uint4*)gB0;
  *(uint4*)&Bs[0][row1 * LDL + kc1] = *(const uint4*)gB1;
  __syncthreads();

  int buf = 0;
  for (int k0 = 0; k0 < H_DIM; k0 += 32, buf ^= 1) {
    const bool more = (k0 + 32 < H_DIM);
    uint4 na0, na1, nb0, nb1;
    if (more) {
      na0 = *(const uint4*)(gA0 + k0 + 32);
      na1 = *(const uint4*)(gA1 + k0 + 32);
      nb0 = *(const uint4*)(gB0 + k0 + 32);
      nb1 = *(const uint4*)(gB1 + k0 + 32);
      if (k0 + 64 < H_DIM) {
        __builtin_prefetch(gA0 + k0 + 64, 0, 3);
        __builtin_prefetch(gB0 + k0 + 64, 0, 3);
      }
    }

    Frag16 a[2], b[4];
#pragma unroll
    for (int s = 0; s < 2; ++s) {
      const int r = wm * 32 + s * 16 + l;
      a[s].q[0] = *(const uint4*)&As[buf][r * LDL + half * 8];
      a[s].q[1] = *(const uint4*)&As[buf][r * LDL + 16 + half * 8];
    }
#pragma unroll
    for (int t = 0; t < 4; ++t) {
      const int n = wn * 64 + t * 16 + l;
      b[t].q[0] = *(const uint4*)&Bs[buf][n * LDL + half * 8];
      b[t].q[1] = *(const uint4*)&Bs[buf][n * LDL + 16 + half * 8];
    }
#pragma unroll
    for (int s = 0; s < 2; ++s)
#pragma unroll
      for (int t = 0; t < 4; ++t) acc[s][t] = wmma_bf16(a[s], b[t], acc[s][t]);

    if (more) {
      *(uint4*)&As[buf ^ 1][row0 * LDL + kc0] = na0;
      *(uint4*)&As[buf ^ 1][row1 * LDL + kc1] = na1;
      *(uint4*)&Bs[buf ^ 1][row0 * LDL + kc0] = nb0;
      *(uint4*)&Bs[buf ^ 1][row1 * LDL + kc1] = nb1;
    }
    __syncthreads();
  }

#pragma unroll
  for (int s = 0; s < 2; ++s) {
#pragma unroll
    for (int t = 0; t < 4; ++t) {
      const int col = nBase + wn * 64 + t * 16 + l;
      const float bb = bo[col];
#pragma unroll
      for (int r = 0; r < 8; ++r) {
        const int m = mBase + wm * 32 + s * 16 + half * 8 + r;
        const size_t off = (size_t)m * H_DIM + col;
        out[off] = acc[s][t][r] + bb + x[off];
      }
    }
  }
}

// ---------------------------------------------------------------------------
// Host launcher
// ---------------------------------------------------------------------------
extern "C" void kernel_launch(void* const* d_in, const int* in_sizes, int n_in,
                              void* d_out, int out_size, void* d_ws,
                              size_t ws_size, hipStream_t stream) {
  const float* x    = (const float*)d_in[0];
  const int*   am   = (const int*)d_in[1];
  const float* ln_g = (const float*)d_in[2];
  const float* ln_b = (const float*)d_in[3];
  const float* Wq   = (const float*)d_in[4];
  const float* bq   = (const float*)d_in[5];
  const float* Wk   = (const float*)d_in[6];
  const float* bk   = (const float*)d_in[7];
  const float* Wv   = (const float*)d_in[8];
  const float* bv   = (const float*)d_in[9];
  const float* Wo   = (const float*)d_in[10];
  const float* bo   = (const float*)d_in[11];
  float* out = (float*)d_out;

  char* ws = (char*)d_ws;
  const size_t NELT = (size_t)M_TOT * H_DIM;           // 8M elements
  unsigned short* h_bf  = (unsigned short*)(ws);                     // 16 MB
  unsigned short* q_bf  = (unsigned short*)(ws + 2 * NELT);          // 16 MB
  unsigned short* k_bf  = (unsigned short*)(ws + 4 * NELT);          // 16 MB
  unsigned short* v_bf  = (unsigned short*)(ws + 6 * NELT);          // 16 MB
  unsigned short* c_bf  = (unsigned short*)(ws + 8 * NELT);          // 16 MB
  unsigned short* w3_bf = (unsigned short*)(ws + 10 * NELT);         //  6 MB
  unsigned short* wo_bf = (unsigned short*)(ws + 10 * NELT + 3u * 2 * 1024 * 1024);

  const size_t WELT = (size_t)H_DIM * H_DIM;           // 1M elements / weight

  // 1) weight conversion fp32 -> bf16 (Wq|Wk|Wv concatenated for one GEMM)
  dim3 cvtGrid((unsigned)(WELT / 4 / 256));
  cvt_bf16_kernel<<<cvtGrid, 256, 0, stream>>>(Wq, w3_bf + 0 * WELT);
  cvt_bf16_kernel<<<cvtGrid, 256, 0, stream>>>(Wk, w3_bf + 1 * WELT);
  cvt_bf16_kernel<<<cvtGrid, 256, 0, stream>>>(Wv, w3_bf + 2 * WELT);
  cvt_bf16_kernel<<<cvtGrid, 256, 0, stream>>>(Wo, wo_bf);

  // 2) LayerNorm -> h (bf16)
  ln_bf16_kernel<<<M_TOT, 256, 0, stream>>>(x, ln_g, ln_b, h_bf);

  // 3) fused QKV projection GEMM (N = 3072)
  gemm_qkv_kernel<<<dim3(3 * H_DIM / 128, M_TOT / 128), 256, 0, stream>>>(
      h_bf, w3_bf, bq, bk, bv, q_bf, k_bf, v_bf);

  // 4) flash attention
  attn_kernel<<<dim3(SEQ / 128, BATCH * NH), 256, 0, stream>>>(
      q_bf, k_bf, v_bf, am, c_bf);

  // 5) output projection + bias + residual (fp32 out)
  gemm_out_kernel<<<dim3(H_DIM / 128, M_TOT / 128), 256, 0, stream>>>(
      c_bf, wo_bf, bo, x, out);
}
